// SelfAttentionBlock_39994735460678
// MI455X (gfx1250) — compile-verified, matrix operand order checked by reading, not run on hardware
//
#include <hip/hip_runtime.h>
#include <hip/hip_bf16.h>

// MI455X / gfx1250: wave32, WMMA 16x16x32 bf16 (f32 accumulate),
// async global->LDS staging (ASYNCcnt) for the projection GEMM.

typedef __attribute__((ext_vector_type(16))) __bf16 v16bf;
typedef __attribute__((ext_vector_type(8)))  float  v8f;

union V16 {              // one WMMA 16-bit operand = 8 VGPRs = 2 x 16 bytes
    v16bf v;
    uint4 q[2];
    unsigned short us[16];
};

#define LSEQ 1024
#define MDIM 64
#define HEADS 8
#define DH 8
#define NH_TOT 64          // 64 logical heads = chunk*8 + batch
#define RSQRT8 0.35355339059327373f
#define NEGBIG -1.0e30f

__device__ __forceinline__ unsigned short bf_bits(float f) {
    return __builtin_bit_cast(unsigned short, (__bf16)f);   // hardware RNE cvt
}
__device__ __forceinline__ float bf16_to_f32(unsigned short h) {
    unsigned u = ((unsigned)h) << 16;
    return __builtin_bit_cast(float, u);
}

// ---------------------------------------------------------------------------
// Kernel 1: projections P = W @ x[b]  (W in {Q,K,V}, [1024,1024] x [1024,64])
// Block = 128 threads (4 waves), one (16-l tile, b, W). Wave w owns m-tile w.
// Per K-step of 32: the x[b, j..j+31, 0..63] slab is DMA'd into LDS with
// GLOBAL_LOAD_ASYNC_TO_LDS_B128 (no VGPR round-trip, ASYNCcnt-tracked),
// overlapped with the A-operand global_load_b128 + v_cvt_pk_bf16_f32 work,
// then fenced with s_wait_asynccnt 0 + barrier. Each wave assembles B from
// the raw f32 slab (strided ds_load_b32 + packed cvt).
// Q/K written bf16 head-layout dst[n][l][d]; V written transposed Vt[n][d][l].
// ---------------------------------------------------------------------------
__global__ __launch_bounds__(128) void proj_heads_kernel(
    const float* __restrict__ x,
    const float* __restrict__ Qw, const float* __restrict__ Kw,
    const float* __restrict__ Vw,
    unsigned short* __restrict__ Qh, unsigned short* __restrict__ Kh,
    unsigned short* __restrict__ Vt)
{
    const int lt = blockIdx.x;           // 16-row tile of output rows (l)
    const int b  = blockIdx.y;           // batch
    const int wz = blockIdx.z;           // which weight matrix
    const float* W = (wz == 0) ? Qw : (wz == 1) ? Kw : Vw;
    unsigned short* dst = (wz == 0) ? Qh : (wz == 1) ? Kh : Vt;

    const int lane  = threadIdx.x & 31;
    const int wave  = threadIdx.x >> 5;  // 4 waves cover m = 0..63
    const int mbase = wave * 16;
    const int lbase = lt * 16;
    const int l15   = lane & 15;
    const int half  = lane >> 4;

    __shared__ __align__(16) float xsr[32 * MDIM];   // raw f32 slab [j][m], 8KB

    // DMA slice per thread: row jj, 16 floats from column mstart
    const int jj     = threadIdx.x >> 2;          // 0..31
    const int mstart = (threadIdx.x & 3) * 16;    // 0,16,32,48
    const float* xsrc = x + (size_t)b * (LSEQ * MDIM) + mstart;
    const unsigned lds_dst = (unsigned)(size_t)&xsr[jj * MDIM + mstart];

    const float* wrow = W + (size_t)(lbase + l15) * LSEQ;   // A row
    const int    mcol = mbase + l15;                        // B col

    v8f acc = {};
    for (int j = 0; j < LSEQ; j += 32) {
        __syncthreads();    // WAR: previous iteration's LDS reads done

        // async DMA: 64 contiguous bytes global -> LDS per thread
        {
            const unsigned long long ga =
                (unsigned long long)(size_t)(xsrc + (size_t)(j + jj) * MDIM);
            asm volatile(
                "global_load_async_to_lds_b128 %0, %1, off\n\t"
                "global_load_async_to_lds_b128 %0, %1, off offset:16\n\t"
                "global_load_async_to_lds_b128 %0, %1, off offset:32\n\t"
                "global_load_async_to_lds_b128 %0, %1, off offset:48"
                :: "v"(lds_dst), "v"(ga) : "memory");
        }

        __builtin_prefetch(wrow + j + 32, 0, 0);   // global_prefetch_b8

        // A (16x32 bf16), overlapped with the async DMA:
        // elem e<8 -> K = half*8+e ; e>=8 -> K = 16+half*8+(e-8)
        V16 A;
        {
            const float* pa = wrow + j + half * 8;
            const float4 f0 = *(const float4*)(pa);
            const float4 f1 = *(const float4*)(pa + 4);
            const float4 f2 = *(const float4*)(pa + 16);
            const float4 f3 = *(const float4*)(pa + 20);
            A.v[0]  = (__bf16)f0.x; A.v[1]  = (__bf16)f0.y;
            A.v[2]  = (__bf16)f0.z; A.v[3]  = (__bf16)f0.w;
            A.v[4]  = (__bf16)f1.x; A.v[5]  = (__bf16)f1.y;
            A.v[6]  = (__bf16)f1.z; A.v[7]  = (__bf16)f1.w;
            A.v[8]  = (__bf16)f2.x; A.v[9]  = (__bf16)f2.y;
            A.v[10] = (__bf16)f2.z; A.v[11] = (__bf16)f2.w;
            A.v[12] = (__bf16)f3.x; A.v[13] = (__bf16)f3.y;
            A.v[14] = (__bf16)f3.z; A.v[15] = (__bf16)f3.w;
        }

        asm volatile("s_wait_asynccnt 0x0" ::: "memory");
        __syncthreads();    // slab visible to all 4 waves

        // B (32x16 bf16): col = mcol, K striping from the raw f32 slab
        V16 Bm;
        {
            const float* pb = &xsr[(half * 8) * MDIM + mcol];
            #pragma unroll
            for (int t = 0; t < 8; ++t) {
                Bm.v[t]     = (__bf16)pb[t * MDIM];          // K = half*8 + t
                Bm.v[t + 8] = (__bf16)pb[(16 + t) * MDIM];   // K = 16+half*8+t
            }
        }
        acc = __builtin_amdgcn_wmma_f32_16x16x32_bf16(false, A.v, false, Bm.v,
                                                      (short)0, acc, false, false);
    }
    // C layout: row = lbase + v + 8*half, col m = mcol.
    // Head mapping: n = (m>>3)*8 + b, d = m&7
    const int n = (mcol >> 3) * HEADS + b;
    const int d = mcol & 7;
    if (wz != 2) {          // Qh/Kh: [n][l][d]
        unsigned short* drow = dst + (size_t)n * LSEQ * DH + d;
        #pragma unroll
        for (int v = 0; v < 8; ++v)
            drow[(size_t)(lbase + v + 8 * half) * DH] = bf_bits(acc[v]);
    } else {                // Vt: [n][d][l]  (transposed for P@V B-operand)
        unsigned short* dcol = dst + ((size_t)n * DH + d) * LSEQ;
        #pragma unroll
        for (int v = 0; v < 8; ++v)
            dcol[lbase + v + 8 * half] = bf_bits(acc[v]);
    }
}

// ---------------------------------------------------------------------------
// Kernel 2: fused flash attention. One wave = one (head n, 16-query tile).
// Streams 32-key tiles: S = Qh*Kh^T (2 WMMA), causal+key-pad mask, online
// softmax, P staged via LDS (C-layout -> A-layout), O += P @ Vt (1 WMMA).
// All WMMA operands assembled with b128 loads.
// ---------------------------------------------------------------------------
__global__ __launch_bounds__(32) void attn_kernel(
    const unsigned short* __restrict__ Qh,
    const unsigned short* __restrict__ Kh,
    const unsigned short* __restrict__ Vt,
    const float* __restrict__ pm,
    const float* __restrict__ x,
    float* __restrict__ out)
{
    const int n  = blockIdx.x;            // 0..63
    const int qt = blockIdx.y;            // 0..63
    const int c  = n >> 3, b = n & 7;
    const int qbase = qt * 16;

    const int lane = threadIdx.x & 31;
    const int l15  = lane & 15;
    const int half = lane >> 4;

    __shared__ __align__(16) unsigned short Pb[16 * 32];  // 16 q x 32 k, bf16

    const uint4 z4 = make_uint4(0u, 0u, 0u, 0u);

    // A_q (16x32, d padded 8->32): lanes<16 carry K=0..7; everything else 0.
    V16 Aq; Aq.q[0] = z4; Aq.q[1] = z4;
    if (half == 0)
        Aq.q[0] = *(const uint4*)(Qh + ((size_t)n * LSEQ + qbase + l15) * DH);

    float mrow[8], srow[8];
    #pragma unroll
    for (int v = 0; v < 8; ++v) { mrow[v] = -3.0e38f; srow[v] = 0.0f; }
    v8f o = {};

    const int nkt = (qt + 2) >> 1;        // ceil((qbase+16)/32)
    for (int kt = 0; kt < nkt; ++kt) {
        const int kbase = kt * 32;

        // B_K tiles (d x key), d padded: lanes<16 carry K(d)=0..7.
        V16 Bk0, Bk1;
        Bk0.q[0] = z4; Bk0.q[1] = z4;
        Bk1.q[0] = z4; Bk1.q[1] = z4;
        if (half == 0) {
            const unsigned short* kr =
                Kh + ((size_t)n * LSEQ + kbase + l15) * DH;
            Bk0.q[0] = *(const uint4*)(kr);
            Bk1.q[0] = *(const uint4*)(kr + 16 * DH);
            __builtin_prefetch(kr + 32 * DH, 0, 0);   // next key tile
        }
        v8f z = {};
        v8f s0 = __builtin_amdgcn_wmma_f32_16x16x32_bf16(false, Aq.v, false, Bk0.v,
                                                         (short)0, z, false, false);
        v8f s1 = __builtin_amdgcn_wmma_f32_16x16x32_bf16(false, Aq.v, false, Bk1.v,
                                                         (short)0, z, false, false);

        const int col0 = kbase + l15, col1 = col0 + 16;
        const float pk0 = pm[b * LSEQ + col0];
        const float pk1 = pm[b * LSEQ + col1];
        float p0[8], p1[8];
        #pragma unroll
        for (int v = 0; v < 8; ++v) {
            const int row = qbase + v + 8 * half;
            float a0 = s0[v] * RSQRT8;
            float a1 = s1[v] * RSQRT8;
            if (col0 > row || pk0 == 0.0f) a0 = NEGBIG;  // causal + key pad
            if (col1 > row || pk1 == 0.0f) a1 = NEGBIG;
            // row max across the 16 lanes holding this row
            float mx = fmaxf(a0, a1);
            mx = fmaxf(mx, __shfl_xor(mx, 1));
            mx = fmaxf(mx, __shfl_xor(mx, 2));
            mx = fmaxf(mx, __shfl_xor(mx, 4));
            mx = fmaxf(mx, __shfl_xor(mx, 8));
            const float mnew = fmaxf(mrow[v], mx);
            const float corr = __expf(mrow[v] - mnew);
            const float e0 = __expf(a0 - mnew);
            const float e1 = __expf(a1 - mnew);
            float ss = e0 + e1;
            ss += __shfl_xor(ss, 1);
            ss += __shfl_xor(ss, 2);
            ss += __shfl_xor(ss, 4);
            ss += __shfl_xor(ss, 8);
            srow[v] = srow[v] * corr + ss;
            mrow[v] = mnew;
            o[v] *= corr;
            p0[v] = e0; p1[v] = e1;
        }

        __syncthreads();   // WAR vs previous iteration's Pb reads
        #pragma unroll
        for (int v = 0; v < 8; ++v) {
            const int prow = v + 8 * half;
            Pb[prow * 32 + l15]      = bf_bits(p0[v]);
            Pb[prow * 32 + l15 + 16] = bf_bits(p1[v]);
        }
        __syncthreads();

        // A_p: re-read P in A-matrix striping (row = l15); two ds_load_b128
        V16 Ap;
        {
            const unsigned short* pr = &Pb[l15 * 32 + half * 8];
            Ap.q[0] = *(const uint4*)(pr);
            Ap.q[1] = *(const uint4*)(pr + 16);
        }
        // B_V (32 keys x d, d padded 8->16): col = l15 (<8 real). Vt is
        // [n][d][L] so each K-chunk is 16 contiguous bytes.
        V16 Bv; Bv.q[0] = z4; Bv.q[1] = z4;
        if (l15 < 8) {
            const unsigned short* vb =
                Vt + ((size_t)n * DH + l15) * LSEQ + kbase + half * 8;
            Bv.q[0] = *(const uint4*)(vb);         // K = half*8 + 0..7
            Bv.q[1] = *(const uint4*)(vb + 16);    // K = 16 + half*8 + 0..7
        }
        o = __builtin_amdgcn_wmma_f32_16x16x32_bf16(false, Ap.v, false, Bv.v,
                                                    (short)0, o, false, false);
    }

    // Normalize, query-side mask, residual, store out[b, q, c*8+d]
    #pragma unroll
    for (int v = 0; v < 8; ++v) {
        const int q = qbase + v + 8 * half;
        if (l15 < 8) {
            const float pmq = pm[b * LSEQ + q];
            float val = (srow[v] > 0.0f) ? (o[v] / srow[v]) : 0.0f;
            val *= pmq;
            const size_t xi = ((size_t)b * LSEQ + q) * MDIM + c * 8 + l15;
            out[xi] = val + x[xi];
        }
    }
}

// ---------------------------------------------------------------------------
// Kernel 3: attention output = mean over c of softmax row q=L-1, * query mask.
// One 256-thread block per batch b; tiny VALU work (64 rows x 1024 keys x d=8).
// ---------------------------------------------------------------------------
__global__ __launch_bounds__(256) void lastrow_attn_kernel(
    const unsigned short* __restrict__ Qh,
    const unsigned short* __restrict__ Kh,
    const float* __restrict__ pm,
    float* __restrict__ attn_out)
{
    const int b   = blockIdx.x;
    const int tid = threadIdx.x;
    __shared__ float red[256];
    __shared__ float qv_s[DH];

    float acc[4] = {0.f, 0.f, 0.f, 0.f};
    const float pmq = pm[b * LSEQ + (LSEQ - 1)];

    for (int cc = 0; cc < 8; ++cc) {
        const int n = cc * HEADS + b;
        if (tid < DH)
            qv_s[tid] = bf16_to_f32(Qh[((size_t)n * LSEQ + (LSEQ - 1)) * DH + tid]);
        __syncthreads();

        float sc[4];
        float lmax = -3.0e38f;
        #pragma unroll
        for (int i = 0; i < 4; ++i) {
            const int k = tid + i * 256;
            const unsigned short* kr = Kh + ((size_t)n * LSEQ + k) * DH;
            float s = 0.0f;
            #pragma unroll
            for (int d = 0; d < DH; ++d) s += qv_s[d] * bf16_to_f32(kr[d]);
            s *= RSQRT8;                                   // causal: k<=1023 ok
            if (pm[b * LSEQ + k] == 0.0f) s = NEGBIG;      // key padding mask
            sc[i] = s;
            lmax = fmaxf(lmax, s);
        }
        red[tid] = lmax; __syncthreads();
        for (int off = 128; off > 0; off >>= 1) {
            if (tid < off) red[tid] = fmaxf(red[tid], red[tid + off]);
            __syncthreads();
        }
        const float Mx = red[0]; __syncthreads();

        float ex[4], lsum = 0.0f;
        #pragma unroll
        for (int i = 0; i < 4; ++i) { ex[i] = __expf(sc[i] - Mx); lsum += ex[i]; }
        red[tid] = lsum; __syncthreads();
        for (int off = 128; off > 0; off >>= 1) {
            if (tid < off) red[tid] += red[tid + off];
            __syncthreads();
        }
        const float Sx = red[0]; __syncthreads();

        #pragma unroll
        for (int i = 0; i < 4; ++i) acc[i] += ex[i] / Sx;
        __syncthreads();   // before qv_s is rewritten next c
    }
    #pragma unroll
    for (int i = 0; i < 4; ++i)
        attn_out[b * LSEQ + tid + i * 256] = acc[i] * pmq * 0.125f;
}

// ---------------------------------------------------------------------------
extern "C" void kernel_launch(void* const* d_in, const int* in_sizes, int n_in,
                              void* d_out, int out_size, void* d_ws, size_t ws_size,
                              hipStream_t stream) {
    const float* x  = (const float*)d_in[0];
    const float* pm = (const float*)d_in[1];
    const float* Qw = (const float*)d_in[2];
    const float* Kw = (const float*)d_in[3];
    const float* Vw = (const float*)d_in[4];
    float* out = (float*)d_out;

    // Workspace: bf16 Qh/Kh [n][l][d], Vt [n][d][l]; 1 MB each (3 MB total)
    unsigned short* Qh = (unsigned short*)d_ws;
    unsigned short* Kh = Qh + (size_t)NH_TOT * LSEQ * DH;
    unsigned short* Vt = Kh + (size_t)NH_TOT * LSEQ * DH;

    proj_heads_kernel<<<dim3(LSEQ / 16, HEADS, 3), 128, 0, stream>>>(
        x, Qw, Kw, Vw, Qh, Kh, Vt);
    attn_kernel<<<dim3(NH_TOT, LSEQ / 16), 32, 0, stream>>>(
        Qh, Kh, Vt, pm, x, out);
    lastrow_attn_kernel<<<HEADS, 256, 0, stream>>>(
        Qh, Kh, pm, out + (size_t)HEADS * LSEQ * MDIM);
}
